// ASNNGoose_37761352466566
// MI455X (gfx1250) — matrix operand order repaired and checked by reading, not compiled, not run on hardware
//
#include <hip/hip_runtime.h>
#include <math.h>

#define DEV __device__ __forceinline__

// ---------- CDNA5 WMMA types ----------
typedef __bf16 bf16_t;
typedef bf16_t v16bf __attribute__((ext_vector_type(16)));
typedef float  v8f   __attribute__((ext_vector_type(8)));
typedef unsigned int u32x4 __attribute__((ext_vector_type(4)));

union FragU { v16bf v; u32x4 q[2]; };

// ---------- model dims ----------
static constexpr int BB  = 4;
static constexpr int TT  = 1024;
static constexpr int DD  = 512;
static constexpr int LL  = 4;
static constexpr int VV  = 32000;
static constexpr int DFF = 2048;
static constexpr int BT  = BB * TT;   // 4096

// ---------- scalar helpers ----------
DEV unsigned short f2bf(float f) {
  union { float f; unsigned u; } c; c.f = f;
  unsigned u = c.u + 0x7FFFu + ((c.u >> 16) & 1u);   // round-to-nearest-even
  return (unsigned short)(u >> 16);
}
DEV float bf2f(unsigned short h) {
  union { unsigned u; float f; } c; c.u = ((unsigned)h) << 16;
  return c.f;
}
DEV float sigmoidf_(float x) { return 1.0f / (1.0f + __expf(-x)); }
DEV float gelu_tanh(float x) {
  float u = 0.7978845608028654f * (x + 0.044715f * x * x * x);
  return 0.5f * x * (1.0f + tanhf(u));
}

// ---------- WMMA GEMM: C(MxN) = A(MxK,bf16,row-major) * Bt(NxK,bf16,row-major)^T ----------
// Block: 256 threads = 8 waves (2 in M x 4 in N). Block tile 64x256.
// Wave tile: 32x64 = 2x4 WMMA(16x16x32) tiles -> 8 independent WMMAs per K-step
// (covers the bf16 WMMA->WMMA RAW hazard window without v_nops).
enum { M_STOREF = 0, M_SPIKE = 1, M_SIG = 2, M_BGELU = 3, M_BADD = 4, M_RMUL = 5 };

template<int MODE>
DEV void gemm_epilogue(v8f acc, int mbase, int nbase, int N, int hsel, int l15,
                       const float* __restrict__ bias, const float* __restrict__ rmul,
                       float* __restrict__ outF, unsigned short* __restrict__ outB,
                       float* __restrict__ xio) {
  const int n = nbase + l15;
#pragma unroll
  for (int r = 0; r < 8; ++r) {
    const int m = mbase + hsel * 8 + r;      // C/D layout: vgpr r -> M=r (lanes 0-15), M=8+r (lanes 16-31)
    const size_t idx = (size_t)m * N + n;
    const float a = acc[r];
    if      (MODE == M_STOREF) outF[idx] = a;
    else if (MODE == M_SPIKE)  outB[idx] = f2bf(a > 0.5f ? 1.0f : (a < -0.5f ? -1.0f : 0.0f));
    else if (MODE == M_SIG)    outF[idx] = sigmoidf_(a);
    else if (MODE == M_BGELU)  outB[idx] = f2bf(gelu_tanh(a + bias[n]));
    else if (MODE == M_BADD)   xio[idx] += a + bias[n];
    else if (MODE == M_RMUL)   xio[idx] += rmul[idx] * a;
  }
}

template<int MODE>
__global__ __launch_bounds__(256) void gemm_wmma_bf16(
    const unsigned short* __restrict__ A,   // (M,K) bf16
    const unsigned short* __restrict__ Bt,  // (N,K) bf16  (pre-transposed B)
    int M, int N, int K,
    const float* __restrict__ bias, const float* __restrict__ rmul,
    float* __restrict__ outF, unsigned short* __restrict__ outB,
    float* __restrict__ xio)
{
  const int lane = threadIdx.x & 31;
  const int wid  = threadIdx.x >> 5;       // 8 waves / block
  const int wm   = wid >> 2;               // 2 waves in M
  const int wn   = wid & 3;                // 4 waves in N
  const int m0   = blockIdx.y * 64  + wm * 32;
  const int n0   = blockIdx.x * 256 + wn * 64;
  const int hsel = lane >> 4;              // which 16-lane half
  const int l15  = lane & 15;

  v8f acc[2][4];
#pragma unroll
  for (int i = 0; i < 2; ++i)
#pragma unroll
    for (int j = 0; j < 4; ++j) acc[i][j] = (v8f){0.f,0.f,0.f,0.f,0.f,0.f,0.f,0.f};

  const unsigned short* a0p = A + (size_t)(m0 + l15)      * K;
  const unsigned short* a1p = A + (size_t)(m0 + 16 + l15) * K;
  const unsigned short* bp[4];
#pragma unroll
  for (int j = 0; j < 4; ++j) bp[j] = Bt + (size_t)(n0 + j * 16 + l15) * K;

  for (int k0 = 0; k0 < K; k0 += 32) {
    if (k0 + 32 < K) {                     // prefetch next K-slab -> global_prefetch_b8
      __builtin_prefetch((const void*)(a0p + k0 + 32), 0, 1);
      __builtin_prefetch((const void*)(a1p + k0 + 32), 0, 1);
#pragma unroll
      for (int j = 0; j < 4; ++j) __builtin_prefetch((const void*)(bp[j] + k0 + 32), 0, 1);
    }
    // A frag (16x32 bf16): lanes 0-15 rows, K = {0..7,16..23}; lanes 16-31 same rows, K = {8..15,24..31}
    FragU a0, a1, bfr[4];
    const int ka = k0 + hsel * 8;
    a0.q[0] = *(const u32x4*)(a0p + ka);
    a0.q[1] = *(const u32x4*)(a0p + ka + 16);
    a1.q[0] = *(const u32x4*)(a1p + ka);
    a1.q[1] = *(const u32x4*)(a1p + ka + 16);
    // B frag (32x16 bf16): lane = N column; lanes 0-15 K = {0..7,16..23}; lanes 16-31 K = {8..15,24..31}
    const int kb = k0 + hsel * 8;
#pragma unroll
    for (int j = 0; j < 4; ++j) {
      bfr[j].q[0] = *(const u32x4*)(bp[j] + kb);
      bfr[j].q[1] = *(const u32x4*)(bp[j] + kb + 16);
    }
    // 8 independent WMMAs; loop-carried dep distance 8 hides the 5-slot bf16 hazard window
#pragma unroll
    for (int j = 0; j < 4; ++j)
      acc[0][j] = __builtin_amdgcn_wmma_f32_16x16x32_bf16(false, a0.v, false, bfr[j].v,
                                                          (short)0, acc[0][j], false, false);
#pragma unroll
    for (int j = 0; j < 4; ++j)
      acc[1][j] = __builtin_amdgcn_wmma_f32_16x16x32_bf16(false, a1.v, false, bfr[j].v,
                                                          (short)0, acc[1][j], false, false);
  }

#pragma unroll
  for (int i = 0; i < 2; ++i)
#pragma unroll
    for (int j = 0; j < 4; ++j)
      gemm_epilogue<MODE>(acc[i][j], m0 + i * 16, n0 + j * 16, N, hsel, l15,
                          bias, rmul, outF, outB, xio);
}

// ---------- LayerNorm (+ optional token-mix producing xk/xv/xr) ----------
DEV float blockReduce256(float v, float* red) {
  const int tid = threadIdx.x;
  __syncthreads();
  red[tid] = v;
  __syncthreads();
#pragma unroll
  for (int off = 128; off > 0; off >>= 1) {
    if (tid < off) red[tid] += red[tid + off];
    __syncthreads();
  }
  return red[0];
}

template<bool MIX>
__global__ __launch_bounds__(256) void ln_mix_kernel(
    const float* __restrict__ x,
    const float* __restrict__ w, const float* __restrict__ b,
    const float* __restrict__ tmk, const float* __restrict__ tmv, const float* __restrict__ tmr,
    unsigned short* __restrict__ ok, unsigned short* __restrict__ ov,
    unsigned short* __restrict__ orr, unsigned short* __restrict__ osingle)
{
  __shared__ float red[256];
  const int tok = blockIdx.x;
  const int t   = tok & (TT - 1);
  const int d0  = threadIdx.x, d1 = threadIdx.x + 256;
  const float* xp = x + (size_t)tok * DD;
  const float a0 = xp[d0], a1 = xp[d1];
  const float s  = blockReduce256(a0 + a1, red);
  const float sq = blockReduce256(a0 * a0 + a1 * a1, red);
  const float mean = s * (1.0f / DD);
  const float var  = sq * (1.0f / DD) - mean * mean;
  const float rstd = rsqrtf(var + 1e-5f);
  const float n0 = (a0 - mean) * rstd * w[d0] + b[d0];
  const float n1 = (a1 - mean) * rstd * w[d1] + b[d1];

  if (!MIX) {
    osingle[(size_t)tok * DD + d0] = f2bf(n0);
    osingle[(size_t)tok * DD + d1] = f2bf(n1);
    return;
  }

  float p0 = 0.0f, p1 = 0.0f;
  if (t > 0) {                     // ln(previous token), zeros at t==0 (uniform per block)
    const float* xq = x + (size_t)(tok - 1) * DD;
    const float c0 = xq[d0], c1 = xq[d1];
    const float s2 = blockReduce256(c0 + c1, red);
    const float q2 = blockReduce256(c0 * c0 + c1 * c1, red);
    const float m2 = s2 * (1.0f / DD);
    const float v2 = q2 * (1.0f / DD) - m2 * m2;
    const float r2 = rsqrtf(v2 + 1e-5f);
    p0 = (c0 - m2) * r2 * w[d0] + b[d0];
    p1 = (c1 - m2) * r2 * w[d1] + b[d1];
  }
  const size_t i0 = (size_t)tok * DD + d0, i1 = (size_t)tok * DD + d1;
  const float tk0 = tmk[d0], tk1 = tmk[d1];
  const float tv0 = tmv[d0], tv1 = tmv[d1];
  const float tr0 = tmr[d0], tr1 = tmr[d1];
  ok [i0] = f2bf(n0 * tk0 + p0 * (1.0f - tk0));  ok [i1] = f2bf(n1 * tk1 + p1 * (1.0f - tk1));
  ov [i0] = f2bf(n0 * tv0 + p0 * (1.0f - tv0));  ov [i1] = f2bf(n1 * tv1 + p1 * (1.0f - tv1));
  orr[i0] = f2bf(n0 * tr0 + p0 * (1.0f - tr0));  orr[i1] = f2bf(n1 * tr1 + p1 * (1.0f - tr1));
}

// ---------- weight transpose + f32->bf16 (LDS-tiled) ----------
__global__ __launch_bounds__(256) void transpose_cvt_bf16(
    const float* __restrict__ in, unsigned short* __restrict__ out, int K, int N)
{
  __shared__ float tile[32][33];
  const size_t lo = (size_t)blockIdx.z * (size_t)K * N;
  const int k0 = blockIdx.y * 32, n0 = blockIdx.x * 32;
  for (int r = threadIdx.y; r < 32; r += 8)
    tile[r][threadIdx.x] = in[lo + (size_t)(k0 + r) * N + n0 + threadIdx.x];
  __syncthreads();
  for (int r = threadIdx.y; r < 32; r += 8)
    out[lo + (size_t)(n0 + r) * K + k0 + threadIdx.x] = f2bf(tile[threadIdx.x][r]);
}

__global__ void cvt_f32_bf16(const float* __restrict__ in, unsigned short* __restrict__ out, int n) {
  const int i = blockIdx.x * blockDim.x + threadIdx.x;
  if (i < n) out[i] = f2bf(in[i]);
}

// ---------- embedding gather ----------
__global__ void embed_kernel(const int* __restrict__ ids, const float* __restrict__ emb,
                             const float* __restrict__ pos, float* __restrict__ x) {
  const int i = blockIdx.x * blockDim.x + threadIdx.x;   // BT*DD
  const int d   = i & (DD - 1);
  const int tok = i >> 9;
  const int t   = tok & (TT - 1);
  x[i] = emb[(size_t)ids[tok] * DD + d] + pos[(size_t)t * DD + d];
}

// ---------- decayed-state scan: S_t = sigmoid(decay_w)*S_{t-1} + k_t*v_t ----------
__global__ void scan_kernel(const unsigned short* __restrict__ kb,
                            const unsigned short* __restrict__ vb,
                            const float* __restrict__ decay_w,
                            unsigned short* __restrict__ Sout) {
  const int i = blockIdx.x * blockDim.x + threadIdx.x;   // BB*DD lanes
  const int d = i & (DD - 1);
  const int b = i >> 9;
  const float dec = sigmoidf_(decay_w[d]);
  float S = 0.0f;
  size_t base = (size_t)b * TT * DD + d;
  for (int t = 0; t < TT; ++t) {
    S = dec * S + bf2f(kb[base]) * bf2f(vb[base]);
    Sout[base] = f2bf(S);
    base += DD;
  }
}

// ---------- host ----------
extern "C" void kernel_launch(void* const* d_in, const int* in_sizes, int n_in,
                              void* d_out, int out_size, void* d_ws, size_t ws_size,
                              hipStream_t stream) {
  const int*   ids     = (const int*)  d_in[0];
  const float* emb     = (const float*)d_in[1];
  const float* pos     = (const float*)d_in[2];
  const float* ln_w    = (const float*)d_in[3];
  const float* ln_b    = (const float*)d_in[4];
  const float* tmk     = (const float*)d_in[5];
  const float* tmv     = (const float*)d_in[6];
  const float* tmr     = (const float*)d_in[7];
  const float* decay_w = (const float*)d_in[8];
  const float* Wk      = (const float*)d_in[9];
  const float* Wv      = (const float*)d_in[10];
  const float* Wr      = (const float*)d_in[11];
  const float* Wo      = (const float*)d_in[12];
  const float* fln_w   = (const float*)d_in[13];
  const float* fln_b   = (const float*)d_in[14];
  const float* W1      = (const float*)d_in[15];
  const float* b1      = (const float*)d_in[16];
  const float* W2      = (const float*)d_in[17];
  const float* b2      = (const float*)d_in[18];
  const float* oln_w   = (const float*)d_in[19];
  const float* oln_b   = (const float*)d_in[20];
  float* logits = (float*)d_out;

  char* ws = (char*)d_ws;
  size_t off = 0;
  auto alloc = [&](size_t bytes) -> void* {
    void* p = ws + off;
    off = (off + bytes + 255) & ~(size_t)255;
    return p;
  };

  float*          x    = (float*)         alloc((size_t)BT * DD * 4);
  unsigned short* xk   = (unsigned short*)alloc((size_t)BT * DD * 2);
  unsigned short* xv   = (unsigned short*)alloc((size_t)BT * DD * 2);
  unsigned short* xr   = (unsigned short*)alloc((size_t)BT * DD * 2);
  unsigned short* kbf  = (unsigned short*)alloc((size_t)BT * DD * 2);
  unsigned short* vbf  = (unsigned short*)alloc((size_t)BT * DD * 2);
  float*          rbuf = (float*)         alloc((size_t)BT * DD * 4);
  unsigned short* Sbuf = (unsigned short*)alloc((size_t)BT * DD * 2);
  unsigned short* xln  = (unsigned short*)alloc((size_t)BT * DD * 2);
  unsigned short* hbuf = (unsigned short*)alloc((size_t)BT * DFF * 2);
  unsigned short* WkT  = (unsigned short*)alloc((size_t)LL * DD * DD * 2);
  unsigned short* WvT  = (unsigned short*)alloc((size_t)LL * DD * DD * 2);
  unsigned short* WrT  = (unsigned short*)alloc((size_t)LL * DD * DD * 2);
  unsigned short* WoT  = (unsigned short*)alloc((size_t)LL * DD * DD * 2);
  unsigned short* W1T  = (unsigned short*)alloc((size_t)LL * DD * DFF * 2);  // (DFF,DD) per layer
  unsigned short* W2T  = (unsigned short*)alloc((size_t)LL * DFF * DD * 2);  // (DD,DFF) per layer
  unsigned short* embB = (unsigned short*)alloc((size_t)VV * DD * 2);        // head B = emb (V rows, D cols)

  const dim3 tb32(32, 8);
  // weight transposes: in (K,N) f32 -> out (N,K) bf16, per-layer via blockIdx.z
  transpose_cvt_bf16<<<dim3(DD / 32,  DD / 32,  LL), tb32, 0, stream>>>(Wk, WkT, DD,  DD);
  transpose_cvt_bf16<<<dim3(DD / 32,  DD / 32,  LL), tb32, 0, stream>>>(Wv, WvT, DD,  DD);
  transpose_cvt_bf16<<<dim3(DD / 32,  DD / 32,  LL), tb32, 0, stream>>>(Wr, WrT, DD,  DD);
  transpose_cvt_bf16<<<dim3(DD / 32,  DD / 32,  LL), tb32, 0, stream>>>(Wo, WoT, DD,  DD);
  transpose_cvt_bf16<<<dim3(DFF / 32, DD / 32,  LL), tb32, 0, stream>>>(W1, W1T, DD,  DFF);
  transpose_cvt_bf16<<<dim3(DD / 32,  DFF / 32, LL), tb32, 0, stream>>>(W2, W2T, DFF, DD);
  cvt_f32_bf16<<<(VV * DD) / 256, 256, 0, stream>>>(emb, embB, VV * DD);

  embed_kernel<<<(BT * DD) / 256, 256, 0, stream>>>(ids, emb, pos, x);

  const dim3 gD  (DD  / 256, BT / 64);   // N=512  -> (2,64)
  const dim3 gF  (DFF / 256, BT / 64);   // N=2048 -> (8,64)
  const dim3 gV  (VV  / 256, BT / 64);   // N=32000-> (125,64)

  for (int l = 0; l < LL; ++l) {
    const size_t wD = (size_t)l * DD * DD;
    const size_t w1 = (size_t)l * DD * DFF;

    ln_mix_kernel<true><<<BT, 256, 0, stream>>>(x, ln_w + l * DD, ln_b + l * DD,
                                                tmk + l * DD, tmv + l * DD, tmr + l * DD,
                                                xk, xv, xr, nullptr);

    gemm_wmma_bf16<M_SPIKE><<<gD, 256, 0, stream>>>(xk, WkT + wD, BT, DD, DD,
                                                    nullptr, nullptr, nullptr, kbf, nullptr);
    gemm_wmma_bf16<M_SPIKE><<<gD, 256, 0, stream>>>(xv, WvT + wD, BT, DD, DD,
                                                    nullptr, nullptr, nullptr, vbf, nullptr);
    gemm_wmma_bf16<M_SIG>  <<<gD, 256, 0, stream>>>(xr, WrT + wD, BT, DD, DD,
                                                    nullptr, nullptr, rbuf, nullptr, nullptr);

    scan_kernel<<<(BB * DD) / 256, 256, 0, stream>>>(kbf, vbf, decay_w + l * DD, Sbuf);

    gemm_wmma_bf16<M_RMUL><<<gD, 256, 0, stream>>>(Sbuf, WoT + wD, BT, DD, DD,
                                                   nullptr, rbuf, nullptr, nullptr, x);

    ln_mix_kernel<false><<<BT, 256, 0, stream>>>(x, fln_w + l * DD, fln_b + l * DD,
                                                 nullptr, nullptr, nullptr,
                                                 nullptr, nullptr, nullptr, xln);

    gemm_wmma_bf16<M_BGELU><<<gF, 256, 0, stream>>>(xln, W1T + w1, BT, DFF, DD,
                                                    b1 + l * DFF, nullptr, nullptr, hbuf, nullptr);
    gemm_wmma_bf16<M_BADD> <<<gD, 256, 0, stream>>>(hbuf, W2T + w1, BT, DD, DFF,
                                                    b2 + l * DD, nullptr, nullptr, nullptr, x);
  }

  ln_mix_kernel<false><<<BT, 256, 0, stream>>>(x, oln_w, oln_b,
                                               nullptr, nullptr, nullptr,
                                               nullptr, nullptr, nullptr, xln);

  // tied head: logits (BT x V) = xln (BT x D) @ emb^T  -> B matrix is embB (V x D) directly
  gemm_wmma_bf16<M_STOREF><<<gV, 256, 0, stream>>>(xln, embB, BT, VV, DD,
                                                   nullptr, nullptr, logits, nullptr, nullptr);
}
